// SequenceTagger_39582418600448
// MI455X (gfx1250) — matrix operand order repaired and checked by reading, not compiled, not run on hardware
//
#include <hip/hip_runtime.h>
#include <hip/hip_bf16.h>

// ---------------------------------------------------------------------------
// Types for WMMA fragments (gfx1250, wave32)
// ---------------------------------------------------------------------------
typedef _Float16 v16h __attribute__((ext_vector_type(16)));
typedef _Float16 v8h  __attribute__((ext_vector_type(8)));
typedef float    v8f  __attribute__((ext_vector_type(8)));

union Frag16 { v16h v; v8h p[2]; };

// A-matrix 16x32 f16 fragment (ISA 7.12.2): lane m = lane&15, h = lane>>4,
// elements 0..7  <-> K = h*8 + e,  elements 8..15 <-> K = 16 + h*8 + (e-8).
__device__ __forceinline__ v16h load_fragA(const _Float16* base, int stride,
                                           int row, int kbase, int h) {
    Frag16 f;
    const _Float16* p = base + (size_t)row * stride + kbase;
    f.p[0] = *reinterpret_cast<const v8h*>(p + h * 8);
    f.p[1] = *reinterpret_cast<const v8h*>(p + 16 + h * 8);
    return f.v;
}

// B-matrix 32x16 f16 fragment: lane n = lane&15, h = lane>>4,
// elements 0..7  <-> K = h*8 + e,  elements 8..15 <-> K = 16 + h*8 + (e-8).
// base is N-major (row = n, K contiguous).
__device__ __forceinline__ v16h load_fragB(const _Float16* base, int stride,
                                           int row, int kbase, int h) {
    Frag16 f;
    const _Float16* p = base + (size_t)row * stride + kbase;
    f.p[0] = *reinterpret_cast<const v8h*>(p + h * 8);
    f.p[1] = *reinterpret_cast<const v8h*>(p + 16 + h * 8);
    return f.v;
}

#define WMMA_F16(A, B, C) \
    __builtin_amdgcn_wmma_f32_16x16x32_f16(false, (A), false, (B), (short)0, (C), false, false)

// ---------------------------------------------------------------------------
// Problem constants
// ---------------------------------------------------------------------------
#define BB 64
#define TT 512
#define EE 128
#define FF 128
#define CC 13
#define UE 16
#define NSHIFT 15          // 1+2+3+4+5 conv taps
#define CTXK   672         // 5*128 + 16, padded to multiple of 32
#define XROWS  68          // 64 + 2-halo each side
#define XLDS_HALF (XROWS * EE)          // 8704 halfs
#define CTXLDS_HALF (64 * CTXK)         // 43008 halfs
#define DYN_LDS ((XLDS_HALF + CTXLDS_HALF) * 2)   // 103424 bytes

// ---------------------------------------------------------------------------
// Kernel 1: pack conv weights (f32 [K,E,F] -> f16 N-major [s][F][E]) and
// dense_w (f32 [656,13] -> f16 padded [16][672]) into workspace.
// ---------------------------------------------------------------------------
__global__ void pack_weights(const float* __restrict__ w1, const float* __restrict__ w2,
                             const float* __restrict__ w3, const float* __restrict__ w4,
                             const float* __restrict__ w5, const float* __restrict__ dense_w,
                             _Float16* __restrict__ Wt, _Float16* __restrict__ Dt) {
    const float* cw[5] = {w1, w2, w3, w4, w5};
    const int totalW = NSHIFT * FF * EE;           // 245760
    const int totalD = 16 * CTXK;                  // 10752
    for (int idx = blockIdx.x * blockDim.x + threadIdx.x;
         idx < totalW + totalD; idx += gridDim.x * blockDim.x) {
        if (idx < totalW) {
            int s = idx >> 14;             // / (128*128)
            int f = (idx >> 7) & 127;
            int e = idx & 127;
            int i, k;
            if      (s < 1)  { i = 0; k = s;      }
            else if (s < 3)  { i = 1; k = s - 1;  }
            else if (s < 6)  { i = 2; k = s - 3;  }
            else if (s < 10) { i = 3; k = s - 6;  }
            else             { i = 4; k = s - 10; }
            // conv_w shape [K,E,F]: flat (k*E + e)*F + f
            Wt[idx] = (_Float16)cw[i][((size_t)k * EE + e) * FF + f];
        } else {
            int j = idx - totalW;
            int n = j / CTXK;
            int k = j % CTXK;
            Dt[j] = (n < CC && k < (5 * FF + UE)) ? (_Float16)dense_w[(size_t)k * CC + n]
                                                  : (_Float16)0.0f;
        }
    }
}

// ---------------------------------------------------------------------------
// Kernel 2: fused gather + 5 convs (WMMA, 2x2 register blocking) + ReLU +
// dense (WMMA) + boundary biases -> potentials [B,T,C] in d_out.
// Grid: (T/64, B), block: 256 threads (8 waves).
// ---------------------------------------------------------------------------
__global__ void __launch_bounds__(256)
conv_dense_pot(const int* __restrict__ input_char, const int* __restrict__ input_script,
               const float* __restrict__ char_emb, const float* __restrict__ uscript_emb,
               const float* __restrict__ cb0, const float* __restrict__ cb1,
               const float* __restrict__ cb2, const float* __restrict__ cb3,
               const float* __restrict__ cb4,
               const float* __restrict__ dense_b, const float* __restrict__ left_b,
               const float* __restrict__ right_b,
               const _Float16* __restrict__ Wt, const _Float16* __restrict__ Dt,
               float* __restrict__ pot) {
    extern __shared__ char smem[];
    _Float16* Xlds   = reinterpret_cast<_Float16*>(smem);                    // [68][128]
    _Float16* CtxLds = reinterpret_cast<_Float16*>(smem + XLDS_HALF * 2);    // [64][672]

    const int b  = blockIdx.y;
    const int t0 = blockIdx.x * 64;
    const int tid  = threadIdx.x;
    const int wave = tid >> 5;
    const int lane = tid & 31;
    const int m16  = lane & 15;
    const int h    = lane >> 4;

    // ---- stage gathered char embeddings (f32 -> f16) with +/-2 halo -------
    for (int idx = tid; idx < XROWS * EE; idx += 256) {
        int r   = idx >> 7;
        int col = idx & 127;
        int t   = t0 + r - 2;
        _Float16 v = (_Float16)0.0f;
        if (t >= 0 && t < TT) {
            int ch = input_char[(size_t)b * TT + t];
            v = (_Float16)char_emb[(size_t)ch * EE + col];
        }
        Xlds[idx] = v;
    }
    // ---- stage script embeddings into ctx cols [640,656) ; zero pad to 672
    for (int idx = tid; idx < 64 * 32; idx += 256) {
        int row = idx >> 5;
        int cl  = idx & 31;
        _Float16 v = (_Float16)0.0f;
        if (cl < UE) {
            int sc = input_script[(size_t)b * TT + t0 + row];
            v = (_Float16)uscript_emb[(size_t)sc * UE + cl];
        }
        CtxLds[row * CTXK + 640 + cl] = v;
    }
    __syncthreads();

    // ---- conv stage: 2x2 tile blocking per wave ---------------------------
    // wave -> (m-pair mp in {0,1} covering mtiles 2mp,2mp+1 ; n-slot g in 0..3)
    // Each wave processes n-tile pair (g*2, g*2+1) of every conv -> 240 WMMAs.
    const float* cbp[5] = {cb0, cb1, cb2, cb3, cb4};
    const int sbase[5]  = {0, 1, 3, 6, 10};
    const int mp = wave >> 2;        // m-pair
    const int g  = wave & 3;         // n-pair slot

    for (int i = 0; i < 5; ++i) {    // conv index, kernel size i+1
        const int Kc    = i + 1;
        const int padlo = i >> 1;    // lax SAME: pad_lo = (K-1)//2
        const int nt0   = i * 8 + g * 2;       // first ctx n-tile of the pair
        const int nrow0 = (g * 2) * 16 + m16;  // row in Wt (N-major) tile 0
        v8f a00 = {0.f,0.f,0.f,0.f,0.f,0.f,0.f,0.f};
        v8f a01 = {0.f,0.f,0.f,0.f,0.f,0.f,0.f,0.f};
        v8f a10 = {0.f,0.f,0.f,0.f,0.f,0.f,0.f,0.f};
        v8f a11 = {0.f,0.f,0.f,0.f,0.f,0.f,0.f,0.f};
        for (int k = 0; k < Kc; ++k) {
            const _Float16* wtile = Wt + (size_t)(sbase[i] + k) * (FF * EE);
            const int rb = (2 * mp) * 16 + k - padlo + 2;   // rows stay in [0,67]
            for (int kc = 0; kc < 4; ++kc) {
                v16h fa0 = load_fragA(Xlds, EE, rb + m16,      kc * 32, h);
                v16h fa1 = load_fragA(Xlds, EE, rb + 16 + m16, kc * 32, h);
                v16h fb0 = load_fragB(wtile, EE, nrow0,      kc * 32, h);
                v16h fb1 = load_fragB(wtile, EE, nrow0 + 16, kc * 32, h);
                a00 = WMMA_F16(fa0, fb0, a00);
                a01 = WMMA_F16(fa0, fb1, a01);
                a10 = WMMA_F16(fa1, fb0, a10);
                a11 = WMMA_F16(fa1, fb1, a11);
            }
        }
        // bias + ReLU, store f16 into ctx LDS (C/D layout: M = r + h*8)
        const float bias0 = cbp[i][g * 32 + m16];
        const float bias1 = cbp[i][g * 32 + 16 + m16];
        const int col0 = nt0 * 16 + m16;
        for (int r = 0; r < 8; ++r) {
            int row0 = (2 * mp) * 16 + h * 8 + r;
            int row1 = row0 + 16;
            CtxLds[row0 * CTXK + col0]      = (_Float16)fmaxf(a00[r] + bias0, 0.0f);
            CtxLds[row0 * CTXK + col0 + 16] = (_Float16)fmaxf(a01[r] + bias1, 0.0f);
            CtxLds[row1 * CTXK + col0]      = (_Float16)fmaxf(a10[r] + bias0, 0.0f);
            CtxLds[row1 * CTXK + col0 + 16] = (_Float16)fmaxf(a11[r] + bias1, 0.0f);
        }
    }
    __syncthreads();

    // ---- dense stage: pot[64,13] = ctx[64,672] @ Dt^T (WMMA, N padded to 16)
    if (wave < 4) {
        int mt = wave;
        v8f acc = {0.f,0.f,0.f,0.f,0.f,0.f,0.f,0.f};
        for (int kc = 0; kc < CTXK / 32; ++kc) {
            v16h fa = load_fragA(CtxLds, CTXK, mt * 16 + m16, kc * 32, h);
            v16h fb = load_fragB(Dt, CTXK, m16, kc * 32, h);
            acc = WMMA_F16(fa, fb, acc);
        }
        int c = m16;
        if (c < CC) {
            float db = dense_b[c], lb = left_b[c], rb = right_b[c];
            for (int r = 0; r < 8; ++r) {
                int m = mt * 16 + h * 8 + r;
                int t = t0 + m;
                float v = acc[r] + db;
                // boundary energies (mask is all-ones in this dispatch)
                if (t == 0)      v += lb;
                if (t == TT - 1) v += rb;
                pot[((size_t)b * TT + t) * CC + c] = v;
            }
        }
    }
}

// ---------------------------------------------------------------------------
// Kernel 3: Viterbi decode, seq_len, trans copy. One wave per batch row.
// ---------------------------------------------------------------------------
__global__ void __launch_bounds__(32)
viterbi(const float* __restrict__ pot, const float* __restrict__ trans,
        const unsigned char* __restrict__ mask,
        int* __restrict__ decoded, int* __restrict__ seq_len,
        float* __restrict__ trans_out) {
    __shared__ unsigned char bp[(TT - 1) * 16];
    __shared__ float st[16];

    const int b = blockIdx.x;
    const int lane = threadIdx.x;
    const int c = lane;

    float tcol[CC];
    if (c < CC)
        for (int p = 0; p < CC; ++p) tcol[p] = trans[p * CC + c];

    float state = (c < CC) ? pot[((size_t)b * TT) * CC + c] : -1e30f;

    for (int t = 1; t < TT; ++t) {
        float best = -1e30f; int arg = 0;
        for (int p = 0; p < CC; ++p) {
            float sp = __shfl(state, p, 32);
            float sc = sp + ((c < CC) ? tcol[p] : 0.0f);
            if (sc > best) { best = sc; arg = p; }
        }
        if (c < CC) {
            bp[(t - 1) * 16 + c] = (unsigned char)arg;
            state = best + pot[((size_t)b * TT + t) * CC + c];
        }
    }
    if (c < CC) st[c] = state;

    // seq_len = popcount(mask[b,:])
    int cnt = 0;
    for (int t = lane; t < TT; t += 32) cnt += mask[(size_t)b * TT + t] ? 1 : 0;
    for (int off = 16; off; off >>= 1) cnt += __shfl_down(cnt, off, 32);
    if (lane == 0) seq_len[b] = cnt;

    __syncthreads();
    if (lane == 0) {
        float best = -1e30f; int last = 0;
        for (int p = 0; p < CC; ++p)
            if (st[p] > best) { best = st[p]; last = p; }
        decoded[(size_t)b * TT + (TT - 1)] = last;
        int tag = last;
        for (int t = TT - 2; t >= 0; --t) {
            tag = bp[t * 16 + tag];
            decoded[(size_t)b * TT + t] = tag;
        }
    }
    if (b == 0)
        for (int idx = lane; idx < CC * CC; idx += 32) trans_out[idx] = trans[idx];
}

// ---------------------------------------------------------------------------
// Host launcher
// ---------------------------------------------------------------------------
extern "C" void kernel_launch(void* const* d_in, const int* in_sizes, int n_in,
                              void* d_out, int out_size, void* d_ws, size_t ws_size,
                              hipStream_t stream) {
    (void)in_sizes; (void)n_in; (void)out_size; (void)ws_size;

    const int*   input_char   = (const int*)d_in[0];
    const int*   input_script = (const int*)d_in[1];
    const unsigned char* input_mask = (const unsigned char*)d_in[2];   // jnp bool: 1 byte
    const float* char_emb     = (const float*)d_in[3];
    const float* uscript_emb  = (const float*)d_in[4];
    const float* dense_w      = (const float*)d_in[5];
    const float* dense_b      = (const float*)d_in[6];
    const float* trans        = (const float*)d_in[7];
    const float* left_b       = (const float*)d_in[8];
    const float* right_b      = (const float*)d_in[9];
    const float* cw[5]; const float* cb[5];
    for (int i = 0; i < 5; ++i) {
        cw[i] = (const float*)d_in[10 + 2 * i];
        cb[i] = (const float*)d_in[11 + 2 * i];
    }

    // workspace: packed f16 weights
    _Float16* Wt = (_Float16*)d_ws;                 // [15][128][128]
    _Float16* Dt = Wt + NSHIFT * FF * EE;           // [16][672]

    // output layout (4-byte elements): decoded | pot | seq_len | trans
    int*   decoded   = (int*)d_out;
    float* pot       = (float*)d_out + BB * TT;
    int*   seqlen    = (int*)d_out + BB * TT + BB * TT * CC;
    float* trans_out = (float*)d_out + BB * TT + BB * TT * CC + BB;

    pack_weights<<<dim3(512), dim3(256), 0, stream>>>(
        cw[0], cw[1], cw[2], cw[3], cw[4], dense_w, Wt, Dt);

    (void)hipFuncSetAttribute(reinterpret_cast<const void*>(conv_dense_pot),
                              hipFuncAttributeMaxDynamicSharedMemorySize, DYN_LDS);
    conv_dense_pot<<<dim3(TT / 64, BB), dim3(256), DYN_LDS, stream>>>(
        input_char, input_script, char_emb, uscript_emb,
        cb[0], cb[1], cb[2], cb[3], cb[4],
        dense_b, left_b, right_b, Wt, Dt, pot);

    viterbi<<<dim3(BB), dim3(32), 0, stream>>>(
        pot, trans, input_mask, decoded, seqlen, trans_out);
}
